// FlashAttention_54812372631763
// MI455X (gfx1250) — matrix operand order repaired and checked by reading, not compiled, not run on hardware
//
#include <hip/hip_runtime.h>

// ---------------------------------------------------------------------------
// Online-softmax P·V (flash-attention second half) for MI455X / gfx1250.
//   O[q, d] = sum_k softmax(S[q,:])_k * V_T[d, k]
// S: [8192, 8192] f32   V_T: [128, 8192] f32   O: [8192, 128] f32
//
// Roofline: S read once = 256 MiB -> ~11.4 us floor at 23.3 TB/s. V_T (4 MiB)
// is L2-resident (192 MB L2). GEMM = 17.2 GFLOP: f32 WMMA (K=4) would be
// compute-bound (~27 us), f16 WMMA (K=32) is ~3-4 us -> memory-bound. So:
// single pass over S, P cast to f16, v_wmma_f32_16x16x32_f16.
//
// CDNA5 data movement: V_T is pre-converted to f16 in d_ws by a prologue
// kernel; the main loop stages 64x128 f16 V tiles into LDS with the Tensor
// Data Mover (tensor_load_to_lds, TENSORcnt-tracked, double-buffered) so the
// DMA overlaps WMMA compute and only one workgroup barrier per K-tile is
// needed. S is software-pipelined in registers (next tile's global_load_b128
// clause issues before the current tile's WMMAs).
// ---------------------------------------------------------------------------

typedef __attribute__((ext_vector_type(16))) _Float16 v16h;
typedef __attribute__((ext_vector_type(8)))  _Float16 v8h;
typedef __attribute__((ext_vector_type(8)))  float    v8f;
typedef __attribute__((ext_vector_type(4)))  float    v4f;
typedef __attribute__((ext_vector_type(4)))  unsigned int v4u;
typedef __attribute__((ext_vector_type(8)))  int      v8i;
typedef __attribute__((ext_vector_type(4)))  int      v4i;

#define SEQ_Q   8192
#define SEQ_K   8192
#define HD      128
#define KT      64              // K-tile per outer iteration (2 WMMA K-steps)
#define WAVES   4
#define RPW     16              // query rows per wave (= WMMA M)
#define RPB     (WAVES * RPW)   // query rows per workgroup

#if __has_builtin(__builtin_amdgcn_tensor_load_to_lds)
#define USE_TDM 1
#else
#define USE_TDM 0
#endif

// ---------------------------------------------------------------------------
// Prologue: V_T f32 -> f16 once (4 MiB -> 2 MiB in d_ws). TDM cannot convert,
// so we feed it pre-converted data; also halves V's L2 traffic.
// ---------------------------------------------------------------------------
__global__ void __launch_bounds__(256)
ConvertV_f16(const float* __restrict__ VT, _Float16* __restrict__ Vh) {
    const size_t i = ((size_t)blockIdx.x * blockDim.x + threadIdx.x) * 8;
    v4f a = *(const v4f*)(VT + i);
    v4f b = *(const v4f*)(VT + i + 4);
    v8h p;
    p[0] = (_Float16)a[0]; p[1] = (_Float16)a[1];
    p[2] = (_Float16)a[2]; p[3] = (_Float16)a[3];
    p[4] = (_Float16)b[0]; p[5] = (_Float16)b[1];
    p[6] = (_Float16)b[2]; p[7] = (_Float16)b[3];
    *(v8h*)(Vh + i) = p;
}

#if USE_TDM
// Issue one TDM descriptor: copy Vh[0..127][kb..kb+63] (f16, row stride 8192)
// into LDS at lds_addr as a contiguous [128][64] f16 tile. D# layout per
// CDNA5 ISA §8.3/§8.4 (2-D tensor: groups 2/3 zero).
__device__ __forceinline__ void tdm_load_v_tile(const _Float16* Vh, int kb,
                                                unsigned int lds_addr) {
    const unsigned long long ga = (unsigned long long)(uintptr_t)(Vh + kb);
    v4u g0;
    g0[0] = 1u;                                   // count=1, no gather
    g0[1] = lds_addr;                             // D#.lds_addr (bytes)
    g0[2] = (unsigned int)ga;                     // global_addr[31:0]
    g0[3] = ((unsigned int)(ga >> 32) & 0x01FFFFFFu) | (2u << 30); // type=2
    v8i g1;
    g1[0] = (int)(1u << 16);                      // wg_mask=0, data_size=2B
    g1[1] = (int)(((unsigned)SEQ_K & 0xFFFFu) << 16);   // tensor_dim0 lo16
    g1[2] = (int)(((unsigned)HD & 0xFFFFu) << 16);      // dim0 hi16=0 | dim1 lo16
    g1[3] = (int)(64u << 16);                     // dim1 hi16=0 | tile_dim0=64
    g1[4] = (int)HD;                              // tile_dim1=128, tile_dim2=0
    g1[5] = (int)SEQ_K;                           // tensor_dim0_stride[31:0]
    g1[6] = 0;                                    // stride0 hi, stride1 lo
    g1[7] = 0;
    v4i g2 = {};
    v4i g3 = {};
#if defined(__clang_major__) && __clang_major__ >= 23
    v8i g4 = {};
    __builtin_amdgcn_tensor_load_to_lds(g0, g1, g2, g3, g4, 0);
#else
    __builtin_amdgcn_tensor_load_to_lds(g0, g1, g2, g3, 0);
#endif
}
#endif

__global__ void __launch_bounds__(WAVES * 32)
FlashPV_wmma_f16(const float* __restrict__ S,
                 const _Float16* __restrict__ Vh,
                 float* __restrict__ O) {
    const int tid  = threadIdx.x;
    const int wave = tid >> 5;
    const int lane = tid & 31;
    const int r    = lane & 15;   // row-within-tile (also N-within-tile for B/D)
    const int h    = lane >> 4;   // which half of the lane pair
    const int qrow = blockIdx.x * RPB + wave * RPW + r;

    __shared__ _Float16 Vt[2][HD][KT];          // 32 KB: double-buffered V tile
    __shared__ _Float16 Pt[WAVES][RPW][KT];     //  8 KB: per-wave P tile
    __shared__ float    Lrow[WAVES][RPW];       // softmax denominators

    v8f acc[8] = {};
    float m    = -3.0e38f;   // running row max
    float lsum = 0.0f;       // per-lane partial row sum (pair-merged at end)

    const float* srow = S + (size_t)qrow * SEQ_K + h * 32;

#if USE_TDM
    if (wave == 0)
        tdm_load_v_tile(Vh, 0, (unsigned int)(uintptr_t)&Vt[0][0][0]);
#endif

    // Prime the S register pipeline with tile 0.
    float scur[32];
    {
        const float* sp = srow;
        #pragma unroll
        for (int j = 0; j < 32; j += 4) {
            v4f t = *(const v4f*)(sp + j);
            scur[j+0] = t[0]; scur[j+1] = t[1];
            scur[j+2] = t[2]; scur[j+3] = t[3];
        }
    }

    int it = 0;
    for (int kb = 0; kb < SEQ_K; kb += KT, ++it) {
        const int cur = it & 1;
        const int nxt = cur ^ 1;

#if USE_TDM
        // TDM(cur) was issued last iteration (or in the prologue): wait for it,
        // publish to all waves, then kick off TDM(nxt) to overlap with compute.
        __builtin_amdgcn_s_wait_tensorcnt(0);
        __syncthreads();
        if (wave == 0 && kb + KT < SEQ_K)
            tdm_load_v_tile(Vh, kb + KT, (unsigned int)(uintptr_t)&Vt[nxt][0][0]);
#else
        // Fallback: cooperative stage of the current V tile (f16 -> f16 copy).
        __syncthreads();
        {
            const v8h* vp = (const v8h*)(Vh + (size_t)tid * SEQ_K + kb);
            #pragma unroll
            for (int j = 0; j < 8; ++j)
                *(v8h*)&Vt[cur][tid][j * 8] = vp[j];
        }
        __syncthreads();
#endif

        // ---- Issue next S tile's loads early (latency hidden under softmax+WMMA).
        float snext[32];
        const bool have_next = (kb + KT) < SEQ_K;
        if (have_next) {
            const float* sp = srow + kb + KT;
            __builtin_prefetch(sp + KT, 0, 1);     // warm L2 two tiles ahead
            #pragma unroll
            for (int j = 0; j < 32; j += 4) {
                v4f t = *(const v4f*)(sp + j);
                snext[j+0] = t[0]; snext[j+1] = t[1];
                snext[j+2] = t[2]; snext[j+3] = t[3];
            }
        }

        // ---- Online softmax update for this row (lane pair r / r+16 = one row).
        float tm = scur[0];
        #pragma unroll
        for (int j = 1; j < 32; ++j) tm = fmaxf(tm, scur[j]);
        tm = fmaxf(tm, __shfl_xor(tm, 16, 32));

        const float mnew  = fmaxf(m, tm);
        const float alpha = __expf(m - mnew);
        m = mnew;
        lsum *= alpha;
        #pragma unroll
        for (int t = 0; t < 8; ++t)
            #pragma unroll
            for (int e = 0; e < 8; ++e) acc[t][e] *= alpha;

        float psum = 0.0f;
        #pragma unroll
        for (int j = 0; j < 32; ++j) {
            float p = __expf(scur[j] - mnew);
            scur[j] = p;
            psum += p;
        }
        lsum += psum;

        // ---- Store P tile (f16) to LDS, row-major [r][k].
        #pragma unroll
        for (int j = 0; j < 32; j += 8) {
            v8h p;
            #pragma unroll
            for (int e = 0; e < 8; ++e) p[e] = (_Float16)scur[j + e];
            *(v8h*)&Pt[wave][r][h * 32 + j] = p;
        }

        // ---- Two K=32 WMMA steps over the K-tile.
        #pragma unroll
        for (int ks = 0; ks < 2; ++ks) {
            // A fragment (16x32 f16): lane m holds K {0..7,16..23},
            // lane m+16 holds K {8..15,24..31} (ISA 16-bit A layout).
            v16h a;
            {
                v8h a0 = *(const v8h*)&Pt[wave][r][ks * 32 + h * 8];
                v8h a1 = *(const v8h*)&Pt[wave][r][ks * 32 + 16 + h * 8];
                #pragma unroll
                for (int e = 0; e < 8; ++e) { a[e] = a0[e]; a[8 + e] = a1[e]; }
            }
            #pragma unroll
            for (int nt = 0; nt < 8; ++nt) {
                // B fragment (32x16 f16): lanes 0-15 hold K {0..7,16..23} of column
                // n=lane; lanes 16-31 hold K {8..15,24..31} (matches A).
                const int n = nt * 16 + r;
                v8h b0 = *(const v8h*)&Vt[cur][n][ks * 32 + h * 8];
                v8h b1 = *(const v8h*)&Vt[cur][n][ks * 32 + 16 + h * 8];
                v16h b;
                #pragma unroll
                for (int e = 0; e < 8; ++e) { b[e] = b0[e]; b[8 + e] = b1[e]; }
                acc[nt] = __builtin_amdgcn_wmma_f32_16x16x32_f16(
                    false, a, false, b, (short)0, acc[nt], false, false);
            }
        }

        // ---- Roll the S register pipeline.
        if (have_next) {
            #pragma unroll
            for (int j = 0; j < 32; ++j) scur[j] = snext[j];
        }
    }

    // ---- Epilogue: merge row sums across lane pairs, normalize, store O.
    lsum += __shfl_xor(lsum, 16, 32);
    if (h == 0) Lrow[wave][r] = lsum;
    __syncthreads();

    const int qbase = blockIdx.x * RPB + wave * RPW;
    #pragma unroll
    for (int v = 0; v < 8; ++v) {
        // D layout: lanes 0-15 -> M=v, lanes 16-31 -> M=8+v; N = lane%16.
        const int   rowoff = h * 8 + v;
        const float linv   = 1.0f / Lrow[wave][rowoff];
        float* orow = O + (size_t)(qbase + rowoff) * HD + r;
        #pragma unroll
        for (int nt = 0; nt < 8; ++nt)
            orow[nt * 16] = acc[nt][v] * linv;
    }
}

extern "C" void kernel_launch(void* const* d_in, const int* in_sizes, int n_in,
                              void* d_out, int out_size, void* d_ws, size_t ws_size,
                              hipStream_t stream) {
    const float* S  = (const float*)d_in[0];
    const float* VT = (const float*)d_in[1];
    float*       O  = (float*)d_out;
    _Float16*    Vh = (_Float16*)d_ws;   // needs 128*8192*2 = 2 MiB of scratch

    // Prologue: V_T f32 -> f16 (1M elements, 8 per thread).
    hipLaunchKernelGGL(ConvertV_f16, dim3((HD * SEQ_K) / (256 * 8)), dim3(256),
                       0, stream, VT, Vh);

    // Main kernel: 128 workgroups x 4 wave32s.
    hipLaunchKernelGGL(FlashPV_wmma_f16, dim3(SEQ_Q / RPB), dim3(WAVES * 32),
                       0, stream, S, Vh, O);
}
